// ProtoNet_85761906966742
// MI455X (gfx1250) — compile-verified, matrix operand order checked by reading, not run on hardware
//
#include <hip/hip_runtime.h>

// Problem constants (reference: N=16384, M=4096, D=512, TEMP=1.0)
#define N_ROWS 16384
#define M_ROWS 4096
#define D_DIM  512

#define KB        32           // K-step (bf16 WMMA K)
#define LDS_PITCH 40           // 32 + 8 pad: keeps b128 alignment, dodges bank conflicts
#define NSTEP     (D_DIM / KB) // 16

typedef __attribute__((ext_vector_type(16))) __bf16 v16bf;
typedef __attribute__((ext_vector_type(8)))  float  v8f;
typedef __attribute__((ext_vector_type(4)))  int    v4i;

typedef __attribute__((address_space(1))) v4i* g128_ptr;   // global int4*
typedef __attribute__((address_space(3))) v4i* l128_ptr;   // LDS int4*

union FragBF {
    v16bf v;
    uint4 q[2];
};

__device__ __forceinline__ unsigned short f2bf_rne(float f) {
    unsigned u = __float_as_uint(f);
    u += 0x7fffu + ((u >> 16) & 1u);   // round-to-nearest-even
    return (unsigned short)(u >> 16);
}

__device__ __forceinline__ uint4 pack8_bf16(float4 a, float4 b) {
    uint4 r;
    r.x = (unsigned)f2bf_rne(a.x) | ((unsigned)f2bf_rne(a.y) << 16);
    r.y = (unsigned)f2bf_rne(a.z) | ((unsigned)f2bf_rne(a.w) << 16);
    r.z = (unsigned)f2bf_rne(b.x) | ((unsigned)f2bf_rne(b.y) << 16);
    r.w = (unsigned)f2bf_rne(b.z) | ((unsigned)f2bf_rne(b.w) << 16);
    return r;
}

// ---- CDNA5 async global->LDS copy (ASYNCcnt-tracked, no VGPR round trip) ----
__device__ __forceinline__ void async_copy_b128(const void* g, void* l) {
#if __has_builtin(__builtin_amdgcn_global_load_async_to_lds_b128)
    __builtin_amdgcn_global_load_async_to_lds_b128((g128_ptr)g, (l128_ptr)l, 0, 0);
#else
    asm volatile("global_load_async_to_lds_b128 %0, %1, off"
                 :: "v"((l128_ptr)l), "v"((g128_ptr)g) : "memory");
#endif
}

__device__ __forceinline__ void wait_asynccnt0() {
#if __has_builtin(__builtin_amdgcn_s_wait_asynccnt)
    __builtin_amdgcn_s_wait_asynccnt(0);
#else
    asm volatile("s_wait_asynccnt 0x0" ::: "memory");
#endif
}

// ---------------------------------------------------------------------------
// Kernel 1: f32 -> bf16 conversion (RNE) + exact f32 row squared-norms.
// One block per row of 512; thread t handles elements t and t+256.
// ---------------------------------------------------------------------------
__global__ __launch_bounds__(256)
void ConvertNormKernel(const float* __restrict__ src,
                       unsigned short* __restrict__ bf,
                       float* __restrict__ sq) {
    __shared__ float red[256];
    const int row = blockIdx.x;
    const int t   = threadIdx.x;
    const float* p = src + (size_t)row * D_DIM;
    float a = p[t];
    float b = p[t + 256];
    unsigned short* q = bf + (size_t)row * D_DIM;
    q[t]       = f2bf_rne(a);
    q[t + 256] = f2bf_rne(b);
    red[t] = a * a + b * b;
    __syncthreads();
    for (int off = 128; off > 0; off >>= 1) {
        if (t < off) red[t] += red[t + off];
        __syncthreads();
    }
    if (t == 0) sq[row] = red[0];
}

// f32-only norm kernel (fallback path)
__global__ __launch_bounds__(256)
void RowSqNormKernel(const float* __restrict__ src,
                     float* __restrict__ dst, int d) {
    __shared__ float red[256];
    const int row = blockIdx.x;
    const float* p = src + (size_t)row * d;
    float s = 0.0f;
    for (int i = threadIdx.x; i < d; i += 256) {
        float v = p[i];
        s += v * v;
    }
    red[threadIdx.x] = s;
    __syncthreads();
    for (int off = 128; off > 0; off >>= 1) {
        if (threadIdx.x < off) red[threadIdx.x] += red[threadIdx.x + off];
        __syncthreads();
    }
    if (threadIdx.x == 0) dst[row] = red[0];
}

// ---------------------------------------------------------------------------
// Shared epilogue: out = t*xy - 0.5*t*(||x||^2 + ||y||^2)
// C/D f32 16x16 layout: VGPR v -> row base+v (lanes 16-31: +8), col = lane%16.
// ---------------------------------------------------------------------------
__device__ __forceinline__ void store_epilogue(v8f acc[2][4],
                                               const float* __restrict__ xsq,
                                               const float* __restrict__ ysq,
                                               float* __restrict__ out,
                                               int n0, int m0, int wr, int wc,
                                               int lane15, int laneHi, float t) {
    const float ht = 0.5f * t;
#pragma unroll
    for (int i = 0; i < 2; ++i) {
        const int rowBase = n0 + wr * 32 + i * 16 + laneHi * 8;
        float xs[8];
#pragma unroll
        for (int v = 0; v < 8; ++v) xs[v] = xsq[rowBase + v];
#pragma unroll
        for (int j = 0; j < 4; ++j) {
            const int col = m0 + wc * 64 + j * 16 + lane15;
            const float ys = ysq[col];
            v8f c = acc[i][j];
#pragma unroll
            for (int v = 0; v < 8; ++v) {
                out[(size_t)(rowBase + v) * M_ROWS + col] = t * c[v] - ht * (xs[v] + ys);
            }
        }
    }
}

// ---------------------------------------------------------------------------
// Kernel 2a (fast path): bf16 WMMA GEMM, double-buffered LDS filled by
// GLOBAL_LOAD_ASYNC_TO_LDS_B128, fused distance epilogue.
// Block = 256 threads (8 wave32s); block tile 128x128; wave tile 32x64.
// ---------------------------------------------------------------------------
__global__ __launch_bounds__(256)
void DistWmmaAsyncKernel(const unsigned short* __restrict__ xbf,
                         const unsigned short* __restrict__ ybf,
                         const float* __restrict__ xsq,
                         const float* __restrict__ ysq,
                         const int*   __restrict__ tempPtr,
                         float* __restrict__ out) {
    __shared__ __align__(16) unsigned short ldsX[2][128 * LDS_PITCH];
    __shared__ __align__(16) unsigned short ldsY[2][128 * LDS_PITCH];

    const int tid  = threadIdx.x;
    const int lane = tid & 31;
    const int wave = tid >> 5;
    const int wr   = wave & 3;
    const int wc   = wave >> 2;

    const int m0 = blockIdx.x * 128;
    const int n0 = blockIdx.y * 128;

    const int lane15 = lane & 15;
    const int laneHi = lane >> 4;
    const int kbase  = laneHi * 8;        // bf16 A/B fragment K split (ISA 7.12.2)

    const float t = (float)tempPtr[0] * 1.0f;

    // Per-thread fill coords: 16B (8 bf16) per async copy; 4 copies per K-step.
    const int fRow = tid >> 2;            // 0..63
    const int fCol = (tid & 3) * 8;       // 0,8,16,24

    auto prefetch = [&](int buf, int kk) {
#pragma unroll
        for (int h = 0; h < 2; ++h) {
            const int r = h * 64 + fRow;
            async_copy_b128(xbf + (size_t)(n0 + r) * D_DIM + kk + fCol,
                            &ldsX[buf][r * LDS_PITCH + fCol]);
            async_copy_b128(ybf + (size_t)(m0 + r) * D_DIM + kk + fCol,
                            &ldsY[buf][r * LDS_PITCH + fCol]);
        }
    };

    v8f acc[2][4];
#pragma unroll
    for (int i = 0; i < 2; ++i)
#pragma unroll
        for (int j = 0; j < 4; ++j)
            acc[i][j] = (v8f)0.0f;

    prefetch(0, 0);

    for (int s = 0; s < NSTEP; ++s) {
        const int cur = s & 1;
        wait_asynccnt0();      // own copies for tile s landed
        __syncthreads();       // everyone's copies landed; everyone done reading buf[cur^1]
        if (s + 1 < NSTEP) prefetch(cur ^ 1, (s + 1) * KB);  // overlaps with compute below

        const unsigned short* bx = ldsX[cur];
        const unsigned short* by = ldsY[cur];

        FragBF a[2], b[4];
#pragma unroll
        for (int i = 0; i < 2; ++i) {
            const int r = wr * 32 + i * 16 + lane15;
            a[i].q[0] = *(const uint4*)&bx[r * LDS_PITCH + kbase];
            a[i].q[1] = *(const uint4*)&bx[r * LDS_PITCH + kbase + 16];
        }
#pragma unroll
        for (int j = 0; j < 4; ++j) {
            const int r = wc * 64 + j * 16 + lane15;
            b[j].q[0] = *(const uint4*)&by[r * LDS_PITCH + kbase];
            b[j].q[1] = *(const uint4*)&by[r * LDS_PITCH + kbase + 16];
        }

#pragma unroll
        for (int i = 0; i < 2; ++i)
#pragma unroll
            for (int j = 0; j < 4; ++j)
                acc[i][j] = __builtin_amdgcn_wmma_f32_16x16x32_bf16(
                    false, a[i].v, false, b[j].v,
                    (short)0, acc[i][j], false, false);
    }

    store_epilogue(acc, xsq, ysq, out, n0, m0, wr, wc, lane15, laneHi, t);
}

// ---------------------------------------------------------------------------
// Kernel 2b (fallback, no workspace needed for bf16 copies): convert f32->bf16
// in-kernel during the LDS fill. Identical math.
// ---------------------------------------------------------------------------
__global__ __launch_bounds__(256)
void DistWmmaKernel(const float* __restrict__ x,
                    const float* __restrict__ y,
                    const float* __restrict__ xsq,
                    const float* __restrict__ ysq,
                    const int*   __restrict__ tempPtr,
                    float* __restrict__ out) {
    __shared__ __align__(16) unsigned short ldsX[128 * LDS_PITCH];
    __shared__ __align__(16) unsigned short ldsY[128 * LDS_PITCH];

    const int tid  = threadIdx.x;
    const int lane = tid & 31;
    const int wave = tid >> 5;
    const int wr   = wave & 3;
    const int wc   = wave >> 2;

    const int m0 = blockIdx.x * 128;
    const int n0 = blockIdx.y * 128;

    const int lane15 = lane & 15;
    const int laneHi = lane >> 4;
    const int kbase  = laneHi * 8;

    const float t = (float)tempPtr[0] * 1.0f;

    v8f acc[2][4];
#pragma unroll
    for (int i = 0; i < 2; ++i)
#pragma unroll
        for (int j = 0; j < 4; ++j)
            acc[i][j] = (v8f)0.0f;

    const int fRow = tid >> 2;
    const int fCol = (tid & 3) * 8;

    for (int kk = 0; kk < D_DIM; kk += KB) {
        __syncthreads();
#pragma unroll
        for (int h = 0; h < 2; ++h) {
            const int r = h * 64 + fRow;
            const float* gp = x + (size_t)(n0 + r) * D_DIM + kk + fCol;
            float4 f0 = *(const float4*)(gp);
            float4 f1 = *(const float4*)(gp + 4);
            *(uint4*)&ldsX[r * LDS_PITCH + fCol] = pack8_bf16(f0, f1);
        }
#pragma unroll
        for (int h = 0; h < 2; ++h) {
            const int r = h * 64 + fRow;
            const float* gp = y + (size_t)(m0 + r) * D_DIM + kk + fCol;
            float4 f0 = *(const float4*)(gp);
            float4 f1 = *(const float4*)(gp + 4);
            *(uint4*)&ldsY[r * LDS_PITCH + fCol] = pack8_bf16(f0, f1);
        }
        __syncthreads();

        FragBF a[2], b[4];
#pragma unroll
        for (int i = 0; i < 2; ++i) {
            const int r = wr * 32 + i * 16 + lane15;
            a[i].q[0] = *(const uint4*)&ldsX[r * LDS_PITCH + kbase];
            a[i].q[1] = *(const uint4*)&ldsX[r * LDS_PITCH + kbase + 16];
        }
#pragma unroll
        for (int j = 0; j < 4; ++j) {
            const int r = wc * 64 + j * 16 + lane15;
            b[j].q[0] = *(const uint4*)&ldsY[r * LDS_PITCH + kbase];
            b[j].q[1] = *(const uint4*)&ldsY[r * LDS_PITCH + kbase + 16];
        }

#pragma unroll
        for (int i = 0; i < 2; ++i)
#pragma unroll
            for (int j = 0; j < 4; ++j)
                acc[i][j] = __builtin_amdgcn_wmma_f32_16x16x32_bf16(
                    false, a[i].v, false, b[j].v,
                    (short)0, acc[i][j], false, false);
    }

    store_epilogue(acc, xsq, ysq, out, n0, m0, wr, wc, lane15, laneHi, t);
}

// ---------------------------------------------------------------------------
extern "C" void kernel_launch(void* const* d_in, const int* in_sizes, int n_in,
                              void* d_out, int out_size, void* d_ws, size_t ws_size,
                              hipStream_t stream) {
    const float* x    = (const float*)d_in[0];
    const float* y    = (const float*)d_in[1];
    const int*   temp = (const int*)d_in[2];
    float*       out  = (float*)d_out;

    // ws layout: xsq[N] f32 | ysq[M] f32 | xbf[N*D] bf16 | ybf[M*D] bf16
    const size_t off_xsq = 0;
    const size_t off_ysq = off_xsq + (size_t)N_ROWS * 4;
    const size_t off_xbf = off_ysq + (size_t)M_ROWS * 4;
    const size_t off_ybf = off_xbf + (size_t)N_ROWS * D_DIM * 2;
    const size_t need    = off_ybf + (size_t)M_ROWS * D_DIM * 2;   // ~20.1 MB

    char* ws = (char*)d_ws;
    float* xsq = (float*)(ws + off_xsq);
    float* ysq = (float*)(ws + off_ysq);

    dim3 grid(M_ROWS / 128, N_ROWS / 128);   // (32, 128)

    if (ws_size >= need) {
        unsigned short* xbf = (unsigned short*)(ws + off_xbf);
        unsigned short* ybf = (unsigned short*)(ws + off_ybf);
        ConvertNormKernel<<<N_ROWS, 256, 0, stream>>>(x, xbf, xsq);
        ConvertNormKernel<<<M_ROWS, 256, 0, stream>>>(y, ybf, ysq);
        DistWmmaAsyncKernel<<<grid, 256, 0, stream>>>(xbf, ybf, xsq, ysq, temp, out);
    } else {
        RowSqNormKernel<<<N_ROWS, 256, 0, stream>>>(x, xsq, D_DIM);
        RowSqNormKernel<<<M_ROWS, 256, 0, stream>>>(y, ysq, D_DIM);
        DistWmmaKernel<<<grid, 256, 0, stream>>>(x, y, xsq, ysq, temp, out);
    }
}
